// ZetaModelII_40295383171263
// MI455X (gfx1250) — matrix operand order repaired and checked by reading, not compiled, not run on hardware
//
#include <hip/hip_runtime.h>
#include <hip/hip_bf16.h>
#include <math.h>

typedef __attribute__((ext_vector_type(16))) _Float16 v16h;
typedef __attribute__((ext_vector_type(8)))  _Float16 v8h;
typedef __attribute__((ext_vector_type(8)))  float    v8f;

#define BT_TOK   8192
#define TLEN     2048
#define BATCH    4
#define DSTATE   64

__device__ __forceinline__ float sigmoidf_(float x){ return 1.f/(1.f+__expf(-x)); }
__device__ __forceinline__ float siluf_(float x){ return x*sigmoidf_(x); }
__device__ __forceinline__ float softplusf_(float x){ return (x>20.f)?x:log1pf(__expf(x)); }

// ---------------------------------------------------------------- max over x[:,:,2]
__global__ void k_max_col2(const float* __restrict__ x, float* __restrict__ ws){
  __shared__ float red[256];
  float m = -1e30f;
  for (int i = threadIdx.x; i < BT_TOK; i += 256) m = fmaxf(m, x[i*4+2]);
  red[threadIdx.x] = m; __syncthreads();
  for (int s = 128; s > 0; s >>= 1){
    if (threadIdx.x < s) red[threadIdx.x] = fmaxf(red[threadIdx.x], red[threadIdx.x+s]);
    __syncthreads();
  }
  if (threadIdx.x == 0) ws[0] = red[0];
}

// ---------------------------------------------------------------- preprocess + lin0 (tanh) -> f16 padded (BT x 32)
__global__ void k_pre_lin0(const float* __restrict__ x, const float* __restrict__ w,
                           const float* __restrict__ b, const float* __restrict__ smc,
                           const float* __restrict__ scal, _Float16* __restrict__ h0h){
  int bt = blockIdx.x*blockDim.x + threadIdx.x;
  if (bt >= BT_TOK) return;
  float start_max = scal[0]/smc[0];
  float xn0 = x[bt*4+0]*(1.f/255.f);
  float xn1 = x[bt*4+1]*(1.f/255.f);
  float xn2 = x[bt*4+2]/start_max;
  float xn3 = x[bt*4+3];
  #pragma unroll
  for (int o = 0; o < 8; ++o){
    float a = b[o] + w[o*4+0]*xn0 + w[o*4+1]*xn1 + w[o*4+2]*xn2 + w[o*4+3]*xn3;
    h0h[bt*32+o] = (_Float16)tanhf(a);
  }
  #pragma unroll
  for (int o = 8; o < 32; ++o) h0h[bt*32+o] = (_Float16)0.f;
}

// ---------------------------------------------------------------- pack f32 weights (N x K) -> zero-padded f16 (Np x Kp)
__global__ void k_pack_w(const float* __restrict__ W, _Float16* __restrict__ Wh,
                         int N, int K, int Np, int Kp){
  int idx = blockIdx.x*blockDim.x + threadIdx.x;
  if (idx >= Np*Kp) return;
  int k = idx % Kp, n = idx / Kp;
  Wh[idx] = (n < N && k < K) ? (_Float16)W[(size_t)n*K + k] : (_Float16)0.f;
}

// ---------------------------------------------------------------- pack f32 activations (M x K) -> zero-padded f16 (M x Kp)
__global__ void k_pack_act(const float* __restrict__ A, _Float16* __restrict__ Ah,
                           int M, int K, int Kp){
  int idx = blockIdx.x*blockDim.x + threadIdx.x;
  if (idx >= M*Kp) return;
  int k = idx % Kp; size_t m = idx / Kp;
  Ah[idx] = (k < K) ? (_Float16)A[m*K + k] : (_Float16)0.f;
}

// ---------------------------------------------------------------- WMMA GEMM: C[M,N] = act(Ah[M,Kp] @ Wh[Np,Kp]^T + bias)
// Padded f16 operands -> branchless inner loop: 4x global_load_b128 + v_wmma per 32-K step.
__global__ void k_wmma_gemm(const _Float16* __restrict__ Ah, const _Float16* __restrict__ Wh,
                            const float* __restrict__ bias, float* __restrict__ C,
                            int M, int N, int Kp, int act){
  int tilesN = (N + 15) >> 4;
  int wave   = blockIdx.x * (blockDim.x >> 5) + (threadIdx.x >> 5);
  int total  = (M >> 4) * tilesN;
  if (wave >= total) return;                 // wave-uniform: EXEC all-ones at WMMA
  int m0 = (wave / tilesN) << 4;
  int n0 = (wave % tilesN) << 4;
  int lane = threadIdx.x & 31;
  int row  = lane & 15;
  int hi   = lane >> 4;

  union { v8f v; float f[8]; } acc;
  #pragma unroll
  for (int i = 0; i < 8; ++i) acc.f[i] = 0.f;

  const _Float16* arow = Ah + (size_t)(m0 + row) * Kp;   // A row M = m0+row
  const _Float16* wrow = Wh + (size_t)(n0 + row) * Kp;   // B col N = n0+row (Np-padded, always valid)

  for (int k0 = 0; k0 < Kp; k0 += 32){
    union { v16h v; v8h h2[2]; } af, bf;
    // A 16x32 layout: lane(0-15) K = {k0..k0+7, k0+16..k0+23}; hi half offset +8
    af.h2[0] = *(const v8h*)(arow + k0 + hi*8);
    af.h2[1] = *(const v8h*)(arow + k0 + 16 + hi*8);
    // B 32x16 layout: lane = N col; lane(0-15) K = {k0..k0+7, k0+16..k0+23}; hi half offset +8
    bf.h2[0] = *(const v8h*)(wrow + k0 + hi*8);
    bf.h2[1] = *(const v8h*)(wrow + k0 + 16 + hi*8);
    acc.v = __builtin_amdgcn_wmma_f32_16x16x32_f16(false, af.v, false, bf.v,
                                                   (short)0, acc.v, false, false);
  }
  int ncol = n0 + row;
  if (ncol >= N) return;
  float bv = bias ? bias[ncol] : 0.f;
  #pragma unroll
  for (int v = 0; v < 8; ++v){               // C/D: lane = N col; VGPR v -> M = v (+8 hi half)
    int m = m0 + hi*8 + v;
    float o = acc.f[v] + bv;
    if (act == 1) o = tanhf(o);
    C[(size_t)m * N + ncol] = o;
  }
}

// ---------------------------------------------------------------- causal depthwise conv + bias + silu (small k)
// writes f32 (BT x di) for the scan AND zero-padded f16 (BT x Kp) for the next GEMM
__global__ void k_dwconv_silu(const float* __restrict__ xz, const float* __restrict__ wconv,
                              const float* __restrict__ bconv, float* __restrict__ xa,
                              _Float16* __restrict__ xah, int di, int kw, int Kp){
  int idx = blockIdx.x*blockDim.x + threadIdx.x;
  if (idx >= BT_TOK*Kp) return;
  int c  = idx % Kp;
  int bt = idx / Kp;
  if (c >= di){ xah[idx] = (_Float16)0.f; return; }
  int b  = bt / TLEN, t = bt % TLEN;
  const float* wr   = wconv + (size_t)c*kw;
  const float* xcol = xz + (size_t)b*TLEN*(2*di) + c;   // xp = first di columns of xz
  float acc = bconv[c];
  for (int j = 0; j < kw; ++j){
    int ts = t - (kw-1) + j;
    if (ts >= 0) acc += wr[j] * xcol[(size_t)ts*(2*di)];
  }
  float s = siluf_(acc);
  xa[(size_t)bt*di + c] = s;
  xah[idx] = (_Float16)s;
}

// ---------------------------------------------------------------- long-tap causal depthwise conv (k=1028): one wave/output
__global__ void k_dwconv_silu_wave(const float* __restrict__ xz, const float* __restrict__ wconv,
                                   const float* __restrict__ bconv, float* __restrict__ xa,
                                   _Float16* __restrict__ xah, int di, int kw, int Kp){
  int wave = blockIdx.x*(blockDim.x >> 5) + (threadIdx.x >> 5);
  int lane = threadIdx.x & 31;
  if (wave >= BT_TOK*Kp) return;
  int c  = wave % Kp;
  int bt = wave / Kp;
  if (c >= di){ if (lane == 0) xah[(size_t)bt*Kp + c] = (_Float16)0.f; return; }
  int b  = bt / TLEN, t = bt % TLEN;
  const float* wr   = wconv + (size_t)c*kw;
  const float* xcol = xz + (size_t)b*TLEN*(2*di) + c;
  float acc = 0.f;
  for (int j = lane; j < kw; j += 32){
    if (j + 256 < kw) __builtin_prefetch(&wr[j + 256], 0, 0);  // global_prefetch_b8
    int ts = t - (kw-1) + j;
    if (ts >= 0) acc += wr[j] * xcol[(size_t)ts*(2*di)];
  }
  #pragma unroll
  for (int off = 16; off; off >>= 1) acc += __shfl_xor(acc, off, 32);
  if (lane == 0){
    float s = siluf_(acc + bconv[c]);
    xa[(size_t)bt*di + c] = s;
    xah[(size_t)bt*Kp + c] = (_Float16)s;
  }
}

// ---------------------------------------------------------------- delta = softplus(dbl[:, :dtr] @ w_dt^T + b_dt)
__global__ void k_delta(const float* __restrict__ dbl, const float* __restrict__ wdt,
                        const float* __restrict__ bdt, float* __restrict__ delta,
                        int di, int dtr, int ldd){
  int idx = blockIdx.x*blockDim.x + threadIdx.x;
  if (idx >= BT_TOK*di) return;
  int d  = idx % di;
  int bt = idx / di;
  float a = bdt[d];
  for (int r = 0; r < dtr; ++r) a += dbl[(size_t)bt*ldd + r] * wdt[d*dtr + r];
  delta[(size_t)bt*di + d] = softplusf_(a);
}

// ---------------------------------------------------------------- selective scan: one wave per (b,d); 2 states/lane
__global__ void k_scan(const float* __restrict__ delta, const float* __restrict__ dbl,
                       const float* __restrict__ xa, const float* __restrict__ Alog,
                       const float* __restrict__ Dp, float* __restrict__ y,
                       int di, int dtr, int ldd){
  int wave = blockIdx.x*(blockDim.x >> 5) + (threadIdx.x >> 5);
  if (wave >= BATCH*di) return;
  int lane = threadIdx.x & 31;
  int b = wave / di, d = wave % di;
  float A0 = -__expf(Alog[(size_t)d*DSTATE + lane]);
  float A1 = -__expf(Alog[(size_t)d*DSTATE + lane + 32]);
  float Dd = Dp[d];
  float h0 = 0.f, h1 = 0.f;
  for (int t = 0; t < TLEN; ++t){
    size_t bt = (size_t)b*TLEN + t;
    float dv = delta[bt*di + d];
    float xv = xa[bt*di + d];
    const float* dr = dbl + bt*ldd + dtr;
    float B0 = dr[lane],          B1 = dr[lane + 32];
    float C0 = dr[DSTATE + lane], C1 = dr[DSTATE + lane + 32];
    float dxb = dv * xv;
    h0 = __expf(dv*A0)*h0 + dxb*B0;
    h1 = __expf(dv*A1)*h1 + dxb*B1;
    float p = h0*C0 + h1*C1;
    #pragma unroll
    for (int off = 16; off; off >>= 1) p += __shfl_xor(p, off, 32);
    if (lane == 0) y[bt*di + d] = p + xv*Dd;
  }
}

// ---------------------------------------------------------------- g = y * silu(z) -> zero-padded f16 (BT x Kp)
__global__ void k_gate(const float* __restrict__ y, const float* __restrict__ xz,
                       _Float16* __restrict__ gh, int di, int Kp){
  int idx = blockIdx.x*blockDim.x + threadIdx.x;
  if (idx >= BT_TOK*Kp) return;
  int d = idx % Kp;
  size_t bt = idx / Kp;
  if (d >= di){ gh[idx] = (_Float16)0.f; return; }
  float z = xz[bt*(2*(size_t)di) + di + d];
  gh[idx] = (_Float16)(y[bt*di + d] * siluf_(z));
}

// ---------------------------------------------------------------- head
__global__ void k_head(const float* __restrict__ m2o,
                       const float* __restrict__ w2, const float* __restrict__ b2,
                       const float* __restrict__ w3, const float* __restrict__ b3,
                       const float* __restrict__ w4, const float* __restrict__ b4,
                       const float* __restrict__ smc, const float* __restrict__ slc,
                       const float* __restrict__ scal, float* __restrict__ out){
  int b = threadIdx.x;
  if (b >= BATCH) return;
  const float* h = m2o + ((size_t)b*TLEN + (TLEN-1))*24;
  float o0[24], o1[24];
  for (int i = 0; i < 24; ++i){
    float a = b2[i];
    for (int j = 0; j < 24; ++j) a += w2[i*24+j]*h[j];
    o0[i] = tanhf(a);
  }
  for (int i = 0; i < 24; ++i){
    float a = b3[i];
    for (int j = 0; j < 24; ++j) a += w3[i*24+j]*o0[j];
    o1[i] = tanhf(a);
  }
  float a = b4[0];
  for (int j = 0; j < 24; ++j) a += w4[j]*o1[j];
  float start_max = scal[0]/smc[0];
  float seq_len   = (float)TLEN/slc[0];
  a *= start_max/seq_len;
  out[b] = (a >= 0.f) ? a : 0.01f*a;
}

// ================================================================ host side
static void launch_gemm(const _Float16* Ah, const _Float16* Wh, const float* bias, float* C,
                        int M, int N, int Kp, int act, hipStream_t stream){
  int tiles  = (M >> 4) * ((N + 15) >> 4);
  int blocks = (tiles + 3) / 4;                 // 4 waves / block
  k_wmma_gemm<<<blocks, 128, 0, stream>>>(Ah, Wh, bias, C, M, N, Kp, act);
}
static void launch_packw(const float* W, _Float16* Wh, int N, int K, int Np, int Kp,
                         hipStream_t stream){
  k_pack_w<<<(Np*Kp + 255)/256, 256, 0, stream>>>(W, Wh, N, K, Np, Kp);
}

extern "C" void kernel_launch(void* const* d_in, const int* in_sizes, int n_in,
                              void* d_out, int out_size, void* d_ws, size_t ws_size,
                              hipStream_t stream){
  (void)in_sizes; (void)n_in; (void)out_size; (void)ws_size;
  const float* x        = (const float*)d_in[0];
  const float* l0w      = (const float*)d_in[1];
  const float* l0b      = (const float*)d_in[2];
  const float* m1_win   = (const float*)d_in[3];   // (96,8)
  const float* m1_wconv = (const float*)d_in[4];   // (48,64)
  const float* m1_bconv = (const float*)d_in[5];
  const float* m1_wx    = (const float*)d_in[6];   // (129,48)
  const float* m1_wdt   = (const float*)d_in[7];   // (48,1)
  const float* m1_bdt   = (const float*)d_in[8];
  const float* m1_Alog  = (const float*)d_in[9];   // (48,64)
  const float* m1_D     = (const float*)d_in[10];
  const float* m1_wout  = (const float*)d_in[11];  // (8,48)
  const float* l1w      = (const float*)d_in[12];  // (24,8)
  const float* l1b      = (const float*)d_in[13];
  const float* m2_win   = (const float*)d_in[14];  // (288,24)
  const float* m2_wconv = (const float*)d_in[15];  // (144,1028)
  const float* m2_bconv = (const float*)d_in[16];
  const float* m2_wx    = (const float*)d_in[17];  // (130,144)
  const float* m2_wdt   = (const float*)d_in[18];  // (144,2)
  const float* m2_bdt   = (const float*)d_in[19];
  const float* m2_Alog  = (const float*)d_in[20];  // (144,64)
  const float* m2_D     = (const float*)d_in[21];
  const float* m2_wout  = (const float*)d_in[22];  // (24,144)
  const float* l2w      = (const float*)d_in[23];
  const float* l2b      = (const float*)d_in[24];
  const float* l3w      = (const float*)d_in[25];
  const float* l3b      = (const float*)d_in[26];
  const float* l4w      = (const float*)d_in[27];
  const float* l4b      = (const float*)d_in[28];
  const float* smc      = (const float*)d_in[29];
  const float* slc      = (const float*)d_in[30];
  float* out = (float*)d_out;

  // ---- bump allocator over d_ws, 256B aligned chunks
  char*  base = (char*)d_ws;
  size_t off  = 0;
  auto alloc = [&](size_t bytes) -> char* {
    off = (off + 255) & ~(size_t)255;
    char* p = base + off; off += bytes; return p;
  };
  float* scal   = (float*)alloc(64*sizeof(float));
  // f32 intermediates
  float* xz1    = (float*)alloc((size_t)BT_TOK*96*4);
  float* xa1    = (float*)alloc((size_t)BT_TOK*48*4);
  float* dbl1   = (float*)alloc((size_t)BT_TOK*129*4);
  float* delta1 = (float*)alloc((size_t)BT_TOK*48*4);
  float* y1     = (float*)alloc((size_t)BT_TOK*48*4);
  float* m1o    = (float*)alloc((size_t)BT_TOK*8*4);
  float* h1     = (float*)alloc((size_t)BT_TOK*24*4);
  float* xz2    = (float*)alloc((size_t)BT_TOK*288*4);
  float* xa2    = (float*)alloc((size_t)BT_TOK*144*4);
  float* dbl2   = (float*)alloc((size_t)BT_TOK*130*4);
  float* delta2 = (float*)alloc((size_t)BT_TOK*144*4);
  float* y2     = (float*)alloc((size_t)BT_TOK*144*4);
  float* m2o    = (float*)alloc((size_t)BT_TOK*24*4);
  // f16 padded GEMM operands (activations)
  _Float16* h0h  = (_Float16*)alloc((size_t)BT_TOK*32*2);
  _Float16* xa1h = (_Float16*)alloc((size_t)BT_TOK*64*2);
  _Float16* g1h  = (_Float16*)alloc((size_t)BT_TOK*64*2);
  _Float16* m1oh = (_Float16*)alloc((size_t)BT_TOK*32*2);
  _Float16* h1h  = (_Float16*)alloc((size_t)BT_TOK*32*2);
  _Float16* xa2h = (_Float16*)alloc((size_t)BT_TOK*160*2);
  _Float16* g2h  = (_Float16*)alloc((size_t)BT_TOK*160*2);
  // f16 padded weights
  _Float16* w1inH  = (_Float16*)alloc((size_t)96*32*2);
  _Float16* wx1H   = (_Float16*)alloc((size_t)144*64*2);
  _Float16* wout1H = (_Float16*)alloc((size_t)16*64*2);
  _Float16* l1wH   = (_Float16*)alloc((size_t)32*32*2);
  _Float16* w2inH  = (_Float16*)alloc((size_t)288*32*2);
  _Float16* wx2H   = (_Float16*)alloc((size_t)144*160*2);
  _Float16* wout2H = (_Float16*)alloc((size_t)32*160*2);

  // ---- pack weights (tiny, once per call)
  launch_packw(m1_win,  w1inH,   96,   8,  96,  32, stream);
  launch_packw(m1_wx,   wx1H,   129,  48, 144,  64, stream);
  launch_packw(m1_wout, wout1H,   8,  48,  16,  64, stream);
  launch_packw(l1w,     l1wH,    24,   8,  32,  32, stream);
  launch_packw(m2_win,  w2inH,  288,  24, 288,  32, stream);
  launch_packw(m2_wx,   wx2H,   130, 144, 144, 160, stream);
  launch_packw(m2_wout, wout2H,  24, 144,  32, 160, stream);

  // ---- forward pass
  k_max_col2<<<1, 256, 0, stream>>>(x, scal);
  k_pre_lin0<<<BT_TOK/256, 256, 0, stream>>>(x, l0w, l0b, smc, scal, h0h);

  // mamba1 (di=48, dtr=1, k=64)
  launch_gemm(h0h, w1inH, nullptr, xz1, BT_TOK, 96, 32, 0, stream);
  k_dwconv_silu<<<(BT_TOK*64 + 255)/256, 256, 0, stream>>>(xz1, m1_wconv, m1_bconv,
                                                           xa1, xa1h, 48, 64, 64);
  launch_gemm(xa1h, wx1H, nullptr, dbl1, BT_TOK, 129, 64, 0, stream);
  k_delta<<<(BT_TOK*48 + 255)/256, 256, 0, stream>>>(dbl1, m1_wdt, m1_bdt, delta1, 48, 1, 129);
  k_scan<<<(BATCH*48)/2, 64, 0, stream>>>(delta1, dbl1, xa1, m1_Alog, m1_D, y1, 48, 1, 129);
  k_gate<<<(BT_TOK*64 + 255)/256, 256, 0, stream>>>(y1, xz1, g1h, 48, 64);
  launch_gemm(g1h, wout1H, nullptr, m1o, BT_TOK, 8, 64, 0, stream);

  // lin1 (tanh)
  k_pack_act<<<(BT_TOK*32 + 255)/256, 256, 0, stream>>>(m1o, m1oh, BT_TOK, 8, 32);
  launch_gemm(m1oh, l1wH, l1b, h1, BT_TOK, 24, 32, 1, stream);

  // mamba2 (di=144, dtr=2, k=1028)
  k_pack_act<<<(BT_TOK*32 + 255)/256, 256, 0, stream>>>(h1, h1h, BT_TOK, 24, 32);
  launch_gemm(h1h, w2inH, nullptr, xz2, BT_TOK, 288, 32, 0, stream);
  k_dwconv_silu_wave<<<(BT_TOK*160)/8, 256, 0, stream>>>(xz2, m2_wconv, m2_bconv,
                                                         xa2, xa2h, 144, 1028, 160);
  launch_gemm(xa2h, wx2H, nullptr, dbl2, BT_TOK, 130, 160, 0, stream);
  k_delta<<<(BT_TOK*144 + 255)/256, 256, 0, stream>>>(dbl2, m2_wdt, m2_bdt, delta2, 144, 2, 130);
  k_scan<<<(BATCH*144)/2, 64, 0, stream>>>(delta2, dbl2, xa2, m2_Alog, m2_D, y2, 144, 2, 130);
  k_gate<<<(BT_TOK*160 + 255)/256, 256, 0, stream>>>(y2, xz2, g2h, 144, 160);
  launch_gemm(g2h, wout2H, nullptr, m2o, BT_TOK, 24, 160, 0, stream);

  // head
  k_head<<<1, 32, 0, stream>>>(m2o, l2w, l2b, l3w, l3b, l4w, l4b, smc, slc, scal, out);
}
